// MultiHeadAttention_61297773249125
// MI455X (gfx1250) — matrix operand order repaired and checked by reading, not compiled, not run on hardware
//
#include <hip/hip_runtime.h>
#include <hip/hip_bf16.h>

#define BATCH 2
#define SEQ   2048
#define EMB   1024
#define NHEAD 16
#define HDIM  64

typedef __bf16          v16bf  __attribute__((ext_vector_type(16)));
typedef float           v8f    __attribute__((ext_vector_type(8)));
typedef float           f32x4  __attribute__((ext_vector_type(4)));
typedef unsigned short  u16x8  __attribute__((ext_vector_type(8)));

union AFrag {
    v16bf v;
    u16x8 h[2];
    unsigned short u[16];
};

// fp32 -> bf16 bits, round-to-nearest-even
__device__ inline unsigned short f2bf_bits(float f) {
    union { float f; unsigned u; } c; c.f = f;
    unsigned r = c.u + 0x7FFFu + ((c.u >> 16) & 1u);
    return (unsigned short)(r >> 16);
}

// ---------------------------------------------------------------------------
// Elementwise fp32 -> bf16 (memory bound, one pass over X and W matrices).
// ---------------------------------------------------------------------------
__global__ __launch_bounds__(256)
void f32_to_bf16(const float* __restrict__ src, unsigned short* __restrict__ dst) {
    const size_t i = (size_t)(blockIdx.x * blockDim.x + threadIdx.x) * 8;
    f32x4 a = *(const f32x4*)(src + i);
    f32x4 b = *(const f32x4*)(src + i + 4);
    u16x8 o;
#pragma unroll
    for (int j = 0; j < 4; ++j) { o[j] = f2bf_bits(a[j]); o[4 + j] = f2bf_bits(b[j]); }
    *(u16x8*)(dst + i) = o;
}

// ---------------------------------------------------------------------------
// GEMM: out[m,n] = sum_k A[m,k] * W[n,k] + bias[n]    (nn.Linear: x @ W.T + b)
//   A bf16 [M=4096, K=1024], W bf16 [N=1024, K=1024]  (both k-contiguous)
// One wave computes a 32x64 tile.  All 6 fragments of a k-step are loaded
// into distinct registers before the 8-wmma batch (one load clause, staggered
// loadcnt waits instead of wait-0 per wmma).
// MODE 0: bf16 ws [B,H,S,D] (Q,K) | MODE 1: bf16 ws [B,H,D,S] (V^T)
// MODE 2: fp32 row-major [M,N] (output projection)
// ---------------------------------------------------------------------------
template <int MODE>
__global__ __launch_bounds__(256)
void gemm_bf16(const unsigned short* __restrict__ A,
               const unsigned short* __restrict__ W,
               const float* __restrict__ bias,
               void* __restrict__ out) {
    const int K = EMB, N = EMB;
    const int wave  = (blockIdx.x * blockDim.x + threadIdx.x) >> 5;
    const int lane  = threadIdx.x & 31;
    const int laneh = lane & 15;
    const int hi    = lane >> 4;
    const int mt    = (wave >> 4) << 5;   // 128 m-tiles of 32
    const int nt    = (wave & 15) << 6;   // 16 n-strips of 64

    const int kbA = hi << 3;
    const unsigned short* arow0 = A + (size_t)(mt + laneh) * K + kbA;
    const unsigned short* arow1 = arow0 + (size_t)16 * K;
    const unsigned short* wrow  = W + (size_t)(nt + laneh) * K + (hi << 3);

    v8f acc[2][4] = {};
    for (int k0 = 0; k0 < K; k0 += 32) {
        AFrag a[2], b[4];
        a[0].h[0] = *(const u16x8*)(arow0 + k0);
        a[0].h[1] = *(const u16x8*)(arow0 + k0 + 16);
        a[1].h[0] = *(const u16x8*)(arow1 + k0);
        a[1].h[1] = *(const u16x8*)(arow1 + k0 + 16);
#pragma unroll
        for (int t = 0; t < 4; ++t) {
            const unsigned short* wp = wrow + (size_t)t * 16 * K + k0;
            b[t].h[0] = *(const u16x8*)wp;
            b[t].h[1] = *(const u16x8*)(wp + 16);
        }
#pragma unroll
        for (int t = 0; t < 4; ++t) {
            acc[0][t] = __builtin_amdgcn_wmma_f32_16x16x32_bf16(
                false, a[0].v, false, b[t].v, (short)0, acc[0][t], false, false);
            acc[1][t] = __builtin_amdgcn_wmma_f32_16x16x32_bf16(
                false, a[1].v, false, b[t].v, (short)0, acc[1][t], false, false);
        }
    }

    // Epilogue: bias + store
#pragma unroll
    for (int mi = 0; mi < 2; ++mi)
#pragma unroll
        for (int t = 0; t < 4; ++t) {
            const int n = nt + t * 16 + laneh;
            const float bval = bias[n];
#pragma unroll
            for (int r = 0; r < 8; ++r) {
                const int m = mt + mi * 16 + r + hi * 8;
                const float v = acc[mi][t][r] + bval;
                if (MODE == 2) {
                    ((float*)out)[(size_t)m * N + n] = v;
                } else {
                    const int b_  = m / SEQ, s = m % SEQ;
                    const int hh = n >> 6,  dd = n & 63;
                    size_t idx;
                    if (MODE == 0) idx = ((size_t)(b_ * NHEAD + hh) * SEQ + s) * HDIM + dd;
                    else           idx = ((size_t)(b_ * NHEAD + hh) * HDIM + dd) * SEQ + s;
                    ((unsigned short*)out)[idx] = f2bf_bits(v);
                }
            }
        }
}

// ---------------------------------------------------------------------------
// One 32-key flash-attention step.  MASK=true only on the single diagonal
// step; interior steps carry no mask VALU.  Softmax runs in exp2 domain
// (scale*log2e folded into the score scale).
// ---------------------------------------------------------------------------
#define SC2 0.1803368801111244f   // (1/sqrt(64)) * log2(e)

template <bool MASK>
__device__ __forceinline__ void attn_step(
    int kt, int q0, int laneh, int hi,
    const unsigned short* __restrict__ Kh,
    const unsigned short* __restrict__ Vh,
    unsigned short* __restrict__ Pw,
    const AFrag* qf, v8f* acc, float* row_m, float* row_l) {

    // ---- load all 4 K fragments (2 key subtiles x 2 d-chunks) ----
    AFrag kf[2][2];
#pragma unroll
    for (int sub = 0; sub < 2; ++sub)
#pragma unroll
        for (int c = 0; c < 2; ++c) {
            const unsigned short* kp =
                Kh + (size_t)(kt + sub * 16 + laneh) * HDIM + c * 32 + (hi << 3);
            kf[sub][c].h[0] = *(const u16x8*)kp;
            kf[sub][c].h[1] = *(const u16x8*)(kp + 16);
        }

    // ---- scores S = Q K^T (K-dim = D = 64) ----
    v8f s[2];
#pragma unroll
    for (int sub = 0; sub < 2; ++sub) {
        v8f z = {};
        z = __builtin_amdgcn_wmma_f32_16x16x32_bf16(
            false, qf[0].v, false, kf[sub][0].v, (short)0, z, false, false);
        z = __builtin_amdgcn_wmma_f32_16x16x32_bf16(
            false, qf[1].v, false, kf[sub][1].v, (short)0, z, false, false);
        s[sub] = z;
    }

    // ---- scale (exp2 domain) + optional causal mask ----
#pragma unroll
    for (int r = 0; r < 8; ++r) {
        float v0 = s[0][r] * SC2;
        float v1 = s[1][r] * SC2;
        if (MASK) {
            const int qrow = q0 + r + hi * 8;
            const int key0 = kt + laneh;
            if (key0 > qrow)      v0 = -1e30f;
            if (key0 + 16 > qrow) v1 = -1e30f;
        }
        s[0][r] = v0; s[1][r] = v1;
    }

    // ---- online softmax (per C-layout row r), exp2 domain ----
    float escale[8];
#pragma unroll
    for (int r = 0; r < 8; ++r) {
        float mx = fmaxf(s[0][r], s[1][r]);
#pragma unroll
        for (int off = 8; off >= 1; off >>= 1)
            mx = fmaxf(mx, __shfl_xor(mx, off, 16));
        const float nm = fmaxf(row_m[r], mx);
        const float p0 = __builtin_exp2f(s[0][r] - nm);
        const float p1 = __builtin_exp2f(s[1][r] - nm);
        float rs = p0 + p1;
#pragma unroll
        for (int off = 8; off >= 1; off >>= 1)
            rs += __shfl_xor(rs, off, 16);
        const float es = __builtin_exp2f(row_m[r] - nm);
        row_l[r] = row_l[r] * es + rs;
        row_m[r] = nm;
        escale[r] = es;
        s[0][r] = p0; s[1][r] = p1;
    }

    // rescale running accumulators
#pragma unroll
    for (int t = 0; t < 4; ++t)
#pragma unroll
        for (int r = 0; r < 8; ++r) acc[t][r] *= escale[r];

    // ---- P: C-layout -> A-layout via per-wave LDS tile ----
#pragma unroll
    for (int r = 0; r < 8; ++r) {
        const int m = r + hi * 8;
        Pw[m * 32 + laneh]      = f2bf_bits(s[0][r]);
        Pw[m * 32 + 16 + laneh] = f2bf_bits(s[1][r]);
    }
    asm volatile("s_wait_dscnt 0" ::: "memory");   // CDNA5 split DS counter
    AFrag pf;
    {
        const int kb = hi << 3;
        const unsigned short* pr = Pw + laneh * 32;
        pf.h[0] = *(const u16x8*)(pr + kb);
        pf.h[1] = *(const u16x8*)(pr + kb + 16);
    }

    // ---- load all 4 V fragments, then O += P @ V (K-dim = 32 keys) ----
    AFrag vf[4];
#pragma unroll
    for (int t = 0; t < 4; ++t) {
        const unsigned short* vp =
            Vh + (size_t)(t * 16 + laneh) * SEQ + kt + (hi << 3);
        vf[t].h[0] = *(const u16x8*)vp;
        vf[t].h[1] = *(const u16x8*)(vp + 16);
    }
#pragma unroll
    for (int t = 0; t < 4; ++t)
        acc[t] = __builtin_amdgcn_wmma_f32_16x16x32_bf16(
            false, pf.v, false, vf[t].v, (short)0, acc[t], false, false);
}

// ---------------------------------------------------------------------------
// Flash attention: one wave per 16-row query tile per (b,h).
// Q,K bf16 [B,H,S,D]; V bf16 [B,H,D,S]; output bf16 [B,S,E] (head-concat).
// ---------------------------------------------------------------------------
__global__ __launch_bounds__(256)
void flash_attn(const unsigned short* __restrict__ Qws,
                const unsigned short* __restrict__ Kws,
                const unsigned short* __restrict__ VwsT,
                unsigned short* __restrict__ attn) {
    __shared__ unsigned short Pld[8][16 * 32];   // per-wave P staging (bf16)

    const int wave   = (blockIdx.x * blockDim.x + threadIdx.x) >> 5;
    const int wlocal = threadIdx.x >> 5;
    const int lane   = threadIdx.x & 31;
    const int laneh  = lane & 15;
    const int hi     = lane >> 4;

    const int qt = wave & 127;           // SEQ/16 = 128 query tiles
    const int h  = (wave >> 7) & (NHEAD - 1);
    const int b  = wave >> 11;
    const int q0 = qt << 4;

    const size_t headoff = (size_t)(b * NHEAD + h) * SEQ * HDIM;
    const unsigned short* Qh = Qws  + headoff;
    const unsigned short* Kh = Kws  + headoff;
    const unsigned short* Vh = VwsT + headoff;   // [D][S]

    // Q fragments: qf[0] covers d=0..31, qf[1] d=32..63
    AFrag qf[2];
    {
        const unsigned short* qrow = Qh + (size_t)(q0 + laneh) * HDIM;
        const int kb = hi << 3;
#pragma unroll
        for (int f = 0; f < 2; ++f) {
            qf[f].h[0] = *(const u16x8*)(qrow + f * 32 + kb);
            qf[f].h[1] = *(const u16x8*)(qrow + f * 32 + kb + 16);
        }
    }

    v8f acc[4] = {};
    float row_m[8], row_l[8];
#pragma unroll
    for (int r = 0; r < 8; ++r) { row_m[r] = -1e30f; row_l[r] = 0.f; }

    unsigned short* Pw = &Pld[wlocal][0];

    // Interior steps: every key in [kt, kt+31] <= every query (no mask).
    const int nfull = (q0 + 1) >> 5;
    for (int step = 0; step < nfull; ++step) {
        const int kt = step << 5;
        __builtin_prefetch(Kh + (size_t)(kt + 32 + laneh) * HDIM, 0, 3);
        attn_step<false>(kt, q0, laneh, hi, Kh, Vh, Pw, qf, acc, row_m, row_l);
    }
    // Single diagonal step with causal masking.
    attn_step<true>(nfull << 5, q0, laneh, hi, Kh, Vh, Pw, qf, acc, row_m, row_l);

    // ---- normalize + store head-concat bf16 [B,S,E] ----
#pragma unroll
    for (int t = 0; t < 4; ++t)
#pragma unroll
        for (int r = 0; r < 8; ++r) {
            const int m  = r + hi * 8;
            const int ss = q0 + m;
            const int d  = t * 16 + laneh;
            attn[(size_t)(b * SEQ + ss) * EMB + h * HDIM + d] =
                f2bf_bits(acc[t][r] / row_l[r]);
        }
}

// ---------------------------------------------------------------------------
extern "C" void kernel_launch(void* const* d_in, const int* in_sizes, int n_in,
                              void* d_out, int out_size, void* d_ws, size_t ws_size,
                              hipStream_t stream) {
    const float* queries = (const float*)d_in[0];
    const float* keys    = (const float*)d_in[1];
    const float* values  = (const float*)d_in[2];
    // d_in[3] = mask: causal triu — computed analytically in flash_attn
    const float* Wq = (const float*)d_in[4];
    const float* bq = (const float*)d_in[5];
    const float* Wk = (const float*)d_in[6];
    const float* bk = (const float*)d_in[7];
    const float* Wv = (const float*)d_in[8];
    const float* bv = (const float*)d_in[9];
    const float* Wo = (const float*)d_in[10];
    const float* bo = (const float*)d_in[11];

    const size_t NE = (size_t)BATCH * SEQ * EMB;   // 4 Mi activation elements
    const size_t NW = (size_t)EMB * EMB;           // 1 Mi weight elements

    unsigned short* p   = (unsigned short*)d_ws;
    unsigned short* Xq  = p;  p += NE;
    unsigned short* Xk  = p;  p += NE;
    unsigned short* Xv  = p;  p += NE;
    unsigned short* Wqb = p;  p += NW;
    unsigned short* Wkb = p;  p += NW;
    unsigned short* Wvb = p;  p += NW;
    unsigned short* Wob = p;  p += NW;
    unsigned short* Qws = p;  p += NE;   // bf16 Q [B,H,S,D]
    unsigned short* Kws = p;  p += NE;   // bf16 K [B,H,S,D]
    unsigned short* Vws = p;  p += NE;   // bf16 V [B,H,D,S]
    unsigned short* atn = p;  p += NE;   // bf16 attn [B,S,E]

    const dim3 blk(256);
    const dim3 cvtX(NE / 8 / 256);   // 2048 blocks
    const dim3 cvtW(NW / 8 / 256);   //  512 blocks
    const dim3 gemm(256);            // 2048 waves = 128 m-tiles x 16 n-strips
    const dim3 flsh(512);            // 4096 waves = B*H*(S/16) q-tiles

    f32_to_bf16<<<cvtX, blk, 0, stream>>>(queries, Xq);
    f32_to_bf16<<<cvtX, blk, 0, stream>>>(keys,    Xk);
    f32_to_bf16<<<cvtX, blk, 0, stream>>>(values,  Xv);
    f32_to_bf16<<<cvtW, blk, 0, stream>>>(Wq, Wqb);
    f32_to_bf16<<<cvtW, blk, 0, stream>>>(Wk, Wkb);
    f32_to_bf16<<<cvtW, blk, 0, stream>>>(Wv, Wvb);
    f32_to_bf16<<<cvtW, blk, 0, stream>>>(Wo, Wob);

    gemm_bf16<0><<<gemm, blk, 0, stream>>>(Xq, Wqb, bq, Qws);
    gemm_bf16<0><<<gemm, blk, 0, stream>>>(Xk, Wkb, bk, Kws);
    gemm_bf16<1><<<gemm, blk, 0, stream>>>(Xv, Wvb, bv, Vws);

    flash_attn<<<flsh, blk, 0, stream>>>(Qws, Kws, Vws, atn);

    gemm_bf16<2><<<gemm, blk, 0, stream>>>(atn, Wob, bo, d_out);
}
